// EVPG_52381421142303
// MI455X (gfx1250) — matrix operand order repaired and checked by reading, compiled, not run on hardware
//
#include <hip/hip_runtime.h>

#define BATCH 1024
#define NSAMP 128
#define DDIM  1024
#define OBSD  256
#define HDIM  256
#define SIGMA 0.05f

typedef __attribute__((ext_vector_type(16))) __bf16       v16bf;
typedef __attribute__((ext_vector_type(8)))  float        v8f;
typedef __attribute__((ext_vector_type(4)))  unsigned int u32x4;
typedef __attribute__((ext_vector_type(4)))  float        f32x4;

union Frag { v16bf v; u32x4 q[2]; };

__device__ __forceinline__ unsigned short f2bf_bits(float f) {
  unsigned u = __builtin_bit_cast(unsigned, f);
  unsigned r = u + 0x7FFFu + ((u >> 16) & 1u);   // round-to-nearest-even
  return (unsigned short)(r >> 16);
}
__device__ __forceinline__ float bf2f(unsigned short h) {
  unsigned u = ((unsigned)h) << 16;
  return __builtin_bit_cast(float, u);
}

// ---------------- prep kernels ----------------

__global__ void init_misc(const float* __restrict__ Wv1, float* __restrict__ wlast,
                          unsigned int* __restrict__ acc) {
  int t = threadIdx.x;
  if (t < HDIM) wlast[t] = Wv1[OBSD * HDIM + t];   // row 256 of [257 x 256]
  if (t == 0) *acc = 0u;
}

__global__ void convert_bf16(const float* __restrict__ src, unsigned short* __restrict__ dst, int n) {
  int i = blockIdx.x * blockDim.x + threadIdx.x;
  if (i < n) dst[i] = f2bf_bits(src[i]);
}

// dst[c*R + r] = bf16(src[r*C + c])   (store weights as [N x K] for contiguous B fragments)
__global__ void transpose_bf16(const float* __restrict__ src, unsigned short* __restrict__ dst,
                               int R, int C) {
  int i = blockIdx.x * blockDim.x + threadIdx.x;
  if (i < R * C) {
    int r = i / C, c = i - r * C;
    dst[(size_t)c * R + r] = f2bf_bits(src[i]);
  }
}

// ---------------- WMMA GEMM ----------------
// C[M x N] = A[M x K](bf16) * Bt[N x K]^T(bf16) + bias
// MODE 0: tanh -> bf16 out;  MODE 1: raw fp32 out;  MODE 2: bias += s*wlast, tanh -> bf16
template <int MODE, int N, int K>
__global__ __launch_bounds__(128) void gemm_wmma(
    const unsigned short* __restrict__ A, const unsigned short* __restrict__ Bt,
    const float* __restrict__ bias, unsigned short* __restrict__ out_bf16,
    float* __restrict__ out_f32, const float* __restrict__ wlast,
    const unsigned int* __restrict__ accp) {
  const int lane = threadIdx.x & 31;
  const int wave = threadIdx.x >> 5;
  const int hf   = lane >> 4;        // which 16-lane half
  const int r    = lane & 15;
  const int n0   = (blockIdx.x * 4 + wave) * 16;
  const int m0   = blockIdx.y * 16;

  v8f acc = {0.f, 0.f, 0.f, 0.f, 0.f, 0.f, 0.f, 0.f};
  const unsigned short* arow = A  + (size_t)(m0 + r) * K;   // A row m0+r
  const unsigned short* brow = Bt + (size_t)(n0 + r) * K;   // B column n0+r (stored as row of Bt)

#pragma unroll
  for (int k0 = 0; k0 < K; k0 += 32) {
    Frag a, b;
    // 16-bit A 16x32: lanes 0-15 hold K {0..7,16..23}, lanes 16-31 hold K {8..15,24..31}
    const u32x4* pa = reinterpret_cast<const u32x4*>(arow + k0 + hf * 8);
    a.q[0] = pa[0];            // K = k0+8h .. +7
    a.q[1] = pa[2];            // K = k0+8h+16 .. +23
    // 16-bit B 32x16: lanes 0-15 hold K {0..7,16..23} of col n, lanes 16-31 hold K {8..15,24..31}
    const u32x4* pb = reinterpret_cast<const u32x4*>(brow + k0 + hf * 8);
    b.q[0] = pb[0];
    b.q[1] = pb[2];
    acc = __builtin_amdgcn_wmma_f32_16x16x32_bf16(false, a.v, false, b.v,
                                                  (short)0, acc, false, false);
  }

  const int n = n0 + r;
  float extra = bias[n];
  if (MODE == 2) {
    float s = (float)(*accp) * (1.0f / (float)NSAMP);  // approx_argmax scalar
    extra += s * wlast[n];
  }
  const int mb = m0 + hf * 8;   // C/D: VGPR v -> row m0 + v + 8*half, col n0+r
#pragma unroll
  for (int v = 0; v < 8; ++v) {
    float val = acc[v] + extra;
    if (MODE == 1) {
      out_f32[(size_t)(mb + v) * N + n] = val;
    } else {
      out_bf16[(size_t)(mb + v) * N + n] = f2bf_bits(tanhf(val));
    }
  }
}

// ---------------- softmax over rows of [BATCH x DDIM], in place ----------------
__global__ __launch_bounds__(256) void softmax_rows(float* __restrict__ x) {
  __shared__ float red[256];
  float* p = x + (size_t)blockIdx.x * DDIM;
  int t = threadIdx.x;
  float m = -3.402823466e38f;
  for (int i = t; i < DDIM; i += 256) m = fmaxf(m, p[i]);
  red[t] = m; __syncthreads();
  for (int s = 128; s > 0; s >>= 1) { if (t < s) red[t] = fmaxf(red[t], red[t + s]); __syncthreads(); }
  m = red[0]; __syncthreads();
  float sum = 0.f;
  for (int i = t; i < DDIM; i += 256) { float e = __expf(p[i] - m); p[i] = e; sum += e; }
  red[t] = sum; __syncthreads();
  for (int s = 128; s > 0; s >>= 1) { if (t < s) red[t] += red[t + s]; __syncthreads(); }
  float inv = 1.0f / red[0];
  for (int i = t; i < DDIM; i += 256) p[i] *= inv;
}

// ---------------- streaming argmax over 512MB of noise (HBM-bound) ----------------
// one wave32 per (b,n) row; sum of argmax indices -> global u32 accumulator.
// Noise is read once -> non-temporal loads (th:NT) so the 512MB stream doesn't
// evict the heavily reused 4MB probs buffer from the 192MB L2.
__global__ __launch_bounds__(256) void argmax_scan(const float* __restrict__ noise,
                                                   const float* __restrict__ probs,
                                                   unsigned int* __restrict__ acc) {
  __shared__ unsigned int s_acc;
  if (threadIdx.x == 0) s_acc = 0u;
  __syncthreads();

  const int wave = threadIdx.x >> 5;
  const int lane = threadIdx.x & 31;
  const int row  = blockIdx.x * 8 + wave;        // < BATCH*NSAMP
  const int b    = row >> 7;                     // row / NSAMP
  const f32x4* nz = reinterpret_cast<const f32x4*>(noise + (size_t)row * DDIM);
  const f32x4* pr = reinterpret_cast<const f32x4*>(probs + (size_t)b * DDIM);

  float bestv = -3.402823466e38f;
  int   besti = 0;
#pragma unroll
  for (int c = 0; c < 8; ++c) {
    const int i4 = c * 32 + lane;
    f32x4 n4 = __builtin_nontemporal_load(nz + i4);  // global_load_b128 th:NT
    f32x4 p4 = pr[i4];                               // RT: stays L2-resident
    float v0 = p4.x + SIGMA * n4.x;
    float v1 = p4.y + SIGMA * n4.y;
    float v2 = p4.z + SIGMA * n4.z;
    float v3 = p4.w + SIGMA * n4.w;
    const int base = i4 * 4;
    if (v0 > bestv) { bestv = v0; besti = base + 0; }
    if (v1 > bestv) { bestv = v1; besti = base + 1; }
    if (v2 > bestv) { bestv = v2; besti = base + 2; }
    if (v3 > bestv) { bestv = v3; besti = base + 3; }
  }
  // wave32 reduction; ties -> lowest index (matches top_k first-max)
  for (int off = 16; off > 0; off >>= 1) {
    float ov = __shfl_xor(bestv, off);
    int   oi = __shfl_xor(besti, off);
    if (ov > bestv || (ov == bestv && oi < besti)) { bestv = ov; besti = oi; }
  }
  if (lane == 0) atomicAdd(&s_acc, (unsigned int)besti);
  __syncthreads();
  if (threadIdx.x == 0) atomicAdd(acc, s_acc);
}

// ---------------- Q = v @ Wv2 + bv2 ----------------
__global__ __launch_bounds__(256) void q_kernel(const unsigned short* __restrict__ vbuf,
                                                const float* __restrict__ Wv2,
                                                const float* __restrict__ bv2,
                                                float* __restrict__ Q) {
  const int row  = blockIdx.x * 8 + (threadIdx.x >> 5);
  const int lane = threadIdx.x & 31;
  float sum = 0.f;
  for (int h = lane; h < HDIM; h += 32)
    sum += bf2f(vbuf[(size_t)row * HDIM + h]) * Wv2[h];
  for (int off = 16; off > 0; off >>= 1) sum += __shfl_xor(sum, off);
  if (lane == 0) Q[row] = sum + bv2[0];
}

// ---------------- launch ----------------
extern "C" void kernel_launch(void* const* d_in, const int* in_sizes, int n_in,
                              void* d_out, int out_size, void* d_ws, size_t ws_size,
                              hipStream_t stream) {
  const float* obs   = (const float*)d_in[0];
  const float* noise = (const float*)d_in[1];
  const float* W1    = (const float*)d_in[2];
  const float* b1    = (const float*)d_in[3];
  const float* W2    = (const float*)d_in[4];
  const float* b2    = (const float*)d_in[5];
  const float* W3    = (const float*)d_in[6];
  const float* b3    = (const float*)d_in[7];
  const float* Wv1   = (const float*)d_in[8];
  const float* bv1   = (const float*)d_in[9];
  const float* Wv2   = (const float*)d_in[10];
  const float* bv2   = (const float*)d_in[11];
  float* Q = (float*)d_out;

  char* base = (char*)d_ws;
  size_t off = 0;
  auto wsalloc = [&](size_t bytes) -> void* {
    void* p = base + off;
    off = (off + bytes + 255) & ~(size_t)255;
    return p;
  };
  unsigned short* A0   = (unsigned short*)wsalloc((size_t)BATCH * OBSD * 2);
  unsigned short* H1   = (unsigned short*)wsalloc((size_t)BATCH * HDIM * 2);
  unsigned short* H2   = (unsigned short*)wsalloc((size_t)BATCH * HDIM * 2);
  unsigned short* VB   = (unsigned short*)wsalloc((size_t)BATCH * HDIM * 2);
  unsigned short* W1t  = (unsigned short*)wsalloc((size_t)OBSD * HDIM * 2);
  unsigned short* W2t  = (unsigned short*)wsalloc((size_t)HDIM * HDIM * 2);
  unsigned short* W3t  = (unsigned short*)wsalloc((size_t)HDIM * DDIM * 2);
  unsigned short* Wv1t = (unsigned short*)wsalloc((size_t)OBSD * HDIM * 2);
  float*        probs  = (float*)wsalloc((size_t)BATCH * DDIM * 4);
  float*        wlast  = (float*)wsalloc(HDIM * 4);
  unsigned int* acc    = (unsigned int*)wsalloc(256);

  // prep (acc must be re-zeroed every call)
  init_misc<<<1, 256, 0, stream>>>(Wv1, wlast, acc);
  convert_bf16<<<(BATCH * OBSD) / 256, 256, 0, stream>>>(obs, A0, BATCH * OBSD);
  transpose_bf16<<<(OBSD * HDIM) / 256, 256, 0, stream>>>(W1, W1t, OBSD, HDIM);
  transpose_bf16<<<(HDIM * HDIM) / 256, 256, 0, stream>>>(W2, W2t, HDIM, HDIM);
  transpose_bf16<<<(HDIM * DDIM) / 256, 256, 0, stream>>>(W3, W3t, HDIM, DDIM);
  transpose_bf16<<<(OBSD * HDIM) / 256, 256, 0, stream>>>(Wv1, Wv1t, OBSD, HDIM); // rows 0..255

  // policy MLP (WMMA bf16)
  gemm_wmma<0, HDIM, OBSD><<<dim3(HDIM / 64, BATCH / 16), 128, 0, stream>>>(
      A0, W1t, b1, H1, nullptr, nullptr, nullptr);
  gemm_wmma<0, HDIM, HDIM><<<dim3(HDIM / 64, BATCH / 16), 128, 0, stream>>>(
      H1, W2t, b2, H2, nullptr, nullptr, nullptr);
  gemm_wmma<1, DDIM, HDIM><<<dim3(DDIM / 64, BATCH / 16), 128, 0, stream>>>(
      H2, W3t, b3, nullptr, probs, nullptr, nullptr);
  softmax_rows<<<BATCH, 256, 0, stream>>>(probs);

  // HBM-bound perturbed argmax scan (512 MB of noise, NT loads)
  argmax_scan<<<(BATCH * NSAMP) / 8, 256, 0, stream>>>(noise, probs, acc);

  // value MLP with rank-1 scalar correction, then Q
  gemm_wmma<2, HDIM, OBSD><<<dim3(HDIM / 64, BATCH / 16), 128, 0, stream>>>(
      A0, Wv1t, bv1, VB, nullptr, wlast, acc);
  q_kernel<<<BATCH / 8, 256, 0, stream>>>(VB, Wv2, bv2, Q);
}
